// TorchLocalAttention_4758823764714
// MI455X (gfx1250) — matrix operand order repaired and checked by reading, not compiled, not run on hardware
//
#include <hip/hip_runtime.h>

// Problem dims (fixed by the reference): x [4,128,64,64] f32, W* [128,128] f32.
#define N_   4
#define C_   128
#define H_   64
#define W_   64
#define HW_  4096
#define K2_  121          // 11x11 window, pad 5
#define PH_  74           // padded rows:  5 + 64 + 5
#define PW_  80           // padded cols:  5 + 64 + 11 (stride 320B)
#define PPLANE_ (PH_ * PW_)            // floats per padded channel plane
#define PADF_   ((size_t)N_ * C_ * PPLANE_)  // floats per padded buffer

typedef unsigned int  u32x4 __attribute__((ext_vector_type(4)));
typedef __bf16        v16bf __attribute__((ext_vector_type(16)));
typedef float         v8f   __attribute__((ext_vector_type(8)));

union FragBF { u32x4 q[2]; v16bf v; };   // 32 bytes = 8 VGPRs

__device__ __forceinline__ unsigned short f2bf(float f) {
  // round-to-nearest-even f32 -> bf16
  unsigned int u = __float_as_uint(f);
  u += 0x7FFFu + ((u >> 16) & 1u);
  return (unsigned short)(u >> 16);
}

// ---------------------------------------------------------------------------
// Convert x [n][c][p] f32 -> xbT [n][p][c] bf16  (channel-contiguous for WMMA B)
// ---------------------------------------------------------------------------
__global__ __launch_bounds__(256)
void k_convert_x(const float* __restrict__ x, unsigned short* __restrict__ xbT) {
  int idx = blockIdx.x * 256 + threadIdx.x;           // over N*C*HW
  if (idx >= N_ * C_ * HW_) return;
  int p = idx & (HW_ - 1);
  int t = idx >> 12;
  int c = t & (C_ - 1);
  int n = t >> 7;
  xbT[((size_t)(n * HW_ + p)) * C_ + c] = f2bf(x[idx]);
}

// Convert W1|W2|W3 [o][c] f32 -> Wb [3][o][c] bf16 (row-major, K contiguous)
__global__ __launch_bounds__(256)
void k_convert_w(const float* __restrict__ W1, const float* __restrict__ W2,
                 const float* __restrict__ W3, unsigned short* __restrict__ Wb) {
  int idx = blockIdx.x * 256 + threadIdx.x;           // over 3*128*128
  if (idx >= 3 * C_ * C_) return;
  const float* W = (idx < C_ * C_) ? W1 : (idx < 2 * C_ * C_ ? W2 : W3);
  Wb[idx] = f2bf(W[idx & (C_ * C_ - 1)]);
}

// Zero-fill (b128 stores) for the padded phi/g planes.
__global__ __launch_bounds__(256)
void k_zero4(u32x4* __restrict__ buf, int n4) {
  int idx = blockIdx.x * 256 + threadIdx.x;
  if (idx < n4) { u32x4 z = {}; buf[idx] = z; }
}

// ---------------------------------------------------------------------------
// y = W * x per (weight, batch): three 128x128x4096 GEMMs.
// One wave32 computes a 16x16 f32 tile via 4x v_wmma_f32_16x16x32_bf16.
// Fragment layouts used here (contiguous K, same order for A and B):
//   A 16x32 bf16: lane<16 row l, K{0..15}; lane>=16 row l-16,
//                 K{16..31}  -> dwords row*64 + half*8 + {0..7}
//   B 32x16 bf16: lane<16 col l, K{0..15}; lane>=16 col l-16, K{16..31}
//                 -> dwords pix*64 + half*8 + {0..7} (xbT is K-contiguous)
//   D 16x16 f32 : VGPR r, lane l -> (M = 8*half + r, N = l&15)
// theta (wi=0) stores dense [n][c][64*64]; phi/g (wi=1/2) store into the
// zero-padded [n][c][74][80] planes so the window stages need no bounds checks.
// ---------------------------------------------------------------------------
__global__ __launch_bounds__(32)
void k_gemm_wmma(const unsigned short* __restrict__ Wb,   // [3][128][128] bf16
                 const unsigned short* __restrict__ xbT,  // [N][4096][128] bf16
                 float* __restrict__ theta,               // [N][128][4096] f32
                 float* __restrict__ phiP,                // [N][128][74][80] f32
                 float* __restrict__ gP)                  // [N][128][74][80] f32
{
  const int pt   = blockIdx.x;        // pixel tile 0..255
  const int ot   = blockIdx.y;        // out-channel tile 0..7
  const int wi   = blockIdx.z >> 2;   // weight 0..2
  const int n    = blockIdx.z & 3;    // batch 0..3
  const int lane = threadIdx.x;       // 0..31, EXEC all ones (required for WMMA)
  const int half = lane >> 4;
  const int l15  = lane & 15;

  const unsigned int* Wd = (const unsigned int*)Wb + (size_t)wi * (C_ * C_ / 2);
  const unsigned int* Xd = (const unsigned int*)xbT;

  const int arow = ot * 16 + l15;                     // A row (out channel)
  const int p    = pt * 16 + l15;                     // B col (pixel)
  const unsigned int* aBase = Wd + (size_t)arow * (C_ / 2) + half * 8;
  const unsigned int* bBase = Xd + ((size_t)(n * HW_ + p)) * (C_ / 2) + half * 8;

  v8f acc = {};
#pragma unroll
  for (int kk = 0; kk < C_ / 32; ++kk) {              // 4 K-steps of 32
    FragBF a, b;
    a.q[0] = *(const u32x4*)(aBase + kk * 16);
    a.q[1] = *(const u32x4*)(aBase + kk * 16 + 4);
    b.q[0] = *(const u32x4*)(bBase + kk * 16);
    b.q[1] = *(const u32x4*)(bBase + kk * 16 + 4);
    acc = __builtin_amdgcn_wmma_f32_16x16x32_bf16(
        /*neg_a=*/false, a.v, /*neg_b=*/false, b.v,
        /*c_mod=*/(short)0, acc, /*reuse_a=*/false, /*reuse_b=*/false);
  }

  const int o0 = ot * 16 + half * 8;                  // first output channel
  const int h  = p >> 6, w = p & (W_ - 1);
  if (wi == 0) {                                      // theta: dense layout
    float* yb = theta + ((size_t)(n * C_ + o0)) * HW_ + p;
#pragma unroll
    for (int r = 0; r < 8; ++r) yb[(size_t)r * HW_] = acc[r];
  } else {                                            // phi/g: padded planes
    float* buf = (wi == 1) ? phiP : gP;
    float* yb = buf + (((size_t)(n * C_ + o0)) * PH_ + (h + 5)) * PW_ + (w + 5);
#pragma unroll
    for (int r = 0; r < 8; ++r) yb[(size_t)r * PPLANE_] = acc[r];
  }
}

// ---------------------------------------------------------------------------
// score[n][ky*11+dx][p] = sum_c theta[n][c][p] * phiP[n][c][h+ky][w+dx]
// One lane per (n, ky, pixel) with 11 dx accumulators: theta loaded once per
// channel (was 11x), phi taps are 11 consecutive floats, zero branches.
// ---------------------------------------------------------------------------
__global__ __launch_bounds__(256)
void k_score(const float* __restrict__ theta, const float* __restrict__ phiP,
             float* __restrict__ sc) {
  int idx = blockIdx.x * 256 + threadIdx.x;           // over N*11*HW
  if (idx >= N_ * 11 * HW_) return;
  int p  = idx & (HW_ - 1);
  int t  = idx >> 12;
  int ky = t % 11;
  int n  = t / 11;
  int h = p >> 6, w = p & (W_ - 1);

  const float* tp = theta + (size_t)(n * C_) * HW_ + p;
  const float* pp = phiP + ((size_t)(n * C_) * PH_ + (h + ky)) * PW_ + w;

  float acc[11] = {};
  for (int c = 0; c < C_; ++c) {
    float tv = tp[(size_t)c * HW_];
    const float* pr = pp + (size_t)c * PPLANE_;
#pragma unroll
    for (int d = 0; d < 11; ++d) acc[d] = fmaf(tv, pr[d], acc[d]);
  }
  float* so = sc + ((size_t)(n * K2_) + ky * 11) * HW_ + p;
#pragma unroll
  for (int d = 0; d < 11; ++d) so[(size_t)d * HW_] = acc[d];
}

// Softmax over the 121 taps, in place; one lane per (n,pixel).
__global__ __launch_bounds__(256)
void k_softmax(float* __restrict__ sc) {
  int idx = blockIdx.x * 256 + threadIdx.x;           // over N*HW
  if (idx >= N_ * HW_) return;
  int p = idx & (HW_ - 1);
  int n = idx >> 12;
  float* col = sc + (size_t)n * K2_ * HW_ + p;
  float m = -3.402823466e38f;
  for (int k = 0; k < K2_; ++k) m = fmaxf(m, col[(size_t)k * HW_]);
  float sum = 0.0f;
  for (int k = 0; k < K2_; ++k) {
    float e = __expf(col[(size_t)k * HW_] - m);
    col[(size_t)k * HW_] = e;
    sum += e;
  }
  float inv = 1.0f / sum;
  for (int k = 0; k < K2_; ++k) col[(size_t)k * HW_] *= inv;
}

// ---------------------------------------------------------------------------
// out[n][c][p] = sum_k gP[n][c][h+ky][w+dx] * weight[n][k][p]
// One lane per (n, pixel, 4-channel group): per-row weights cached in regs
// (weight traffic /4), g taps contiguous, no bounds checks (padded planes).
// ---------------------------------------------------------------------------
__global__ __launch_bounds__(256)
void k_weight(const float* __restrict__ gP, const float* __restrict__ wgt,
              float* __restrict__ out) {
  int idx = blockIdx.x * 256 + threadIdx.x;           // over N*HW*(C/4)
  if (idx >= N_ * HW_ * (C_ / 4)) return;
  int p  = idx & (HW_ - 1);
  int t  = idx >> 12;
  int cg = t & 31;                                    // channel group 0..31
  int n  = t >> 5;
  int h = p >> 6, w = p & (W_ - 1);

  const float* wb = wgt + (size_t)(n * K2_) * HW_ + p;
  const float* gb = gP + ((size_t)(n * C_ + cg * 4)) * PH_ * PW_;

  float acc[4] = {};
  for (int ky = 0; ky < 11; ++ky) {
    float wk[11];
#pragma unroll
    for (int d = 0; d < 11; ++d) wk[d] = wb[(size_t)(ky * 11 + d) * HW_];
    const float* grow = gb + (size_t)(h + ky) * PW_ + w;
#pragma unroll
    for (int j = 0; j < 4; ++j) {
      const float* gr = grow + (size_t)j * PPLANE_;
#pragma unroll
      for (int d = 0; d < 11; ++d) acc[j] = fmaf(gr[d], wk[d], acc[j]);
    }
  }
  float* ob = out + ((size_t)(n * C_ + cg * 4)) * HW_ + p;
#pragma unroll
  for (int j = 0; j < 4; ++j) ob[(size_t)j * HW_] = acc[j];
}

// ---------------------------------------------------------------------------
extern "C" void kernel_launch(void* const* d_in, const int* in_sizes, int n_in,
                              void* d_out, int out_size, void* d_ws, size_t ws_size,
                              hipStream_t stream) {
  (void)in_sizes; (void)n_in; (void)out_size; (void)ws_size;
  const float* x  = (const float*)d_in[0];
  const float* W1 = (const float*)d_in[1];
  const float* W2 = (const float*)d_in[2];
  const float* W3 = (const float*)d_in[3];
  float* out = (float*)d_out;

  // Workspace carve (all sections 16B aligned):
  char* ws = (char*)d_ws;
  unsigned short* xbT = (unsigned short*)ws; ws += (size_t)N_ * HW_ * C_ * 2;     // 4 MiB
  unsigned short* Wb  = (unsigned short*)ws; ws += (size_t)3 * C_ * C_ * 2;       // 96 KiB
  float* theta = (float*)ws;                 ws += (size_t)N_ * C_ * HW_ * 4;     // 8 MiB
  float* phiP  = (float*)ws;                 ws += PADF_ * 4;                     // 11.6 MiB
  float* gP    = (float*)ws;                 ws += PADF_ * 4;                     // 11.6 MiB (adjacent)
  float* sc    = (float*)ws;                 // N*121*HW*4 = 7.6 MiB

  k_convert_x<<<(N_ * C_ * HW_) / 256, 256, 0, stream>>>(x, xbT);
  k_convert_w<<<(3 * C_ * C_ + 255) / 256, 256, 0, stream>>>(W1, W2, W3, Wb);

  // Zero both padded buffers (contiguous) so halo taps read exact zeros.
  int n4 = (int)(2 * PADF_ / 4);
  k_zero4<<<(n4 + 255) / 256, 256, 0, stream>>>((u32x4*)phiP, n4);

  // theta|phi|g GEMMs fused into one grid (z = wi*4 + n)
  k_gemm_wmma<<<dim3(HW_ / 16, C_ / 16, 3 * N_), 32, 0, stream>>>(
      Wb, xbT, theta, phiP, gP);

  k_score  <<<(N_ * 11 * HW_) / 256, 256, 0, stream>>>(theta, phiP, sc);
  k_softmax<<<(N_ * HW_ + 255) / 256, 256, 0, stream>>>(sc);
  k_weight <<<(N_ * HW_ * (C_ / 4)) / 256, 256, 0, stream>>>(gP, sc, out);
}
